// ODEModel_16887811408631
// MI455X (gfx1250) — matrix operand order repaired and checked by reading, not compiled, not run on hardware
//
#include <hip/hip_runtime.h>
#include <math.h>

// ---------------------------------------------------------------------------
// CDNA5 (gfx1250, wave32) RK4 neural-ODE integrator.
// Layer2 GEMM [256,4096]x[4096,2048] runs on v_wmma_f32_16x16x32_bf16,
// split-K=2, 2 m-tiles x 2 n-tiles per wave (4 independent accumulators).
// Fragment layouts follow CDNA5 ISA §7.12.2:
//   A (16x32 bf16): lane<16 -> M=lane, dword v<4: K=2v,2v+1 ; v>=4: K=16+2(v-4)
//                   lane>=16 -> M=lane-16, K offsets +8 / +24
//   B (32x16 bf16): same K mapping as A: lane<16 -> N=lane, v<4: K=2v,2v+1 ; v>=4: K=16+2(v-4)
//                   lane>=16 -> N=lane-16, K offsets +8 / +24
//   C/D (16x16 f32): VGPR r, lane<16 -> M=r, N=lane ; lane>=16 -> M=8+r, N=lane-16
// ---------------------------------------------------------------------------

typedef __attribute__((ext_vector_type(16))) __bf16 v16bf;
typedef __attribute__((ext_vector_type(8)))  float  v8f;

#define B_SZ     256
#define D_IN     4
#define H1       4096
#define H2       2048
#define NSTEP    16
#define KBLKS    (H1 / 32)      // 128 k-blocks of 32

__device__ __forceinline__ unsigned f2bf(float f) {
    // round-to-nearest-even fp32 -> bf16 (relu outputs: no NaN concern)
    unsigned u = __builtin_bit_cast(unsigned, f);
    return (u + 0x7FFFu + ((u >> 16) & 1u)) >> 16;
}

// ---- one-time: pack W2 (row-major [2048,4096] f32) into bf16 B-fragments ---
// storage: idx = ((n_tile*KBLKS + kblk)*32 + lane)*8 + v   (dwords, 2 bf16 each)
__global__ void pack_w2_kernel(const float* __restrict__ W2,
                               unsigned* __restrict__ Bp) {
    int id = blockIdx.x * 256 + threadIdx.x;          // 0 .. 4194303
    int v      = id & 7;
    int lane   = (id >> 3) & 31;
    int kblk   = (id >> 8) & (KBLKS - 1);
    int n_tile = id >> 15;
    int n  = n_tile * 16 + (lane & 15);
    int k0 = kblk * 32 + ((lane >> 4) ? 8 : 0)
           + ((v < 4) ? 2 * v : 16 + 2 * (v - 4));         // B layout
    unsigned lo = f2bf(W2[n * H1 + k0]);
    unsigned hi = f2bf(W2[n * H1 + k0 + 1]);
    Bp[id] = lo | (hi << 16);
}

// ---- init: y = z = y0, out[0] = y0 ----------------------------------------
__global__ void init_kernel(const float* __restrict__ y0,
                            float* __restrict__ y, float* __restrict__ z,
                            float* __restrict__ out) {
    int i = blockIdx.x * 256 + threadIdx.x;
    if (i < B_SZ * D_IN) { float v = y0[i]; y[i] = v; z[i] = v; out[i] = v; }
}

// ---- layer 1: h1 = relu(z @ W1^T + b1), written straight into A-fragments --
// storage: idx = ((m_tile*KBLKS + kblk)*32 + lane)*8 + v
__global__ void layer1_kernel(const float* __restrict__ z,
                              const float* __restrict__ W1,
                              const float* __restrict__ b1,
                              unsigned* __restrict__ Ap) {
    int id = blockIdx.x * 256 + threadIdx.x;          // 0 .. 524287
    int v      = id & 7;
    int lane   = (id >> 3) & 31;
    int kblk   = (id >> 8) & (KBLKS - 1);
    int m_tile = id >> 15;
    int M  = m_tile * 16 + (lane & 15);
    int K0 = kblk * 32 + ((lane >> 4) ? 8 : 0)
           + ((v < 4) ? 2 * v : 16 + 2 * (v - 4));    // A layout
    float z0 = z[M * 4 + 0], z1 = z[M * 4 + 1], z2 = z[M * 4 + 2], z3 = z[M * 4 + 3];
    const float* w0 = W1 + (size_t)K0 * 4;
    float lo = b1[K0]     + z0 * w0[0] + z1 * w0[1] + z2 * w0[2] + z3 * w0[3];
    float hi = b1[K0 + 1] + z0 * w0[4] + z1 * w0[5] + z2 * w0[6] + z3 * w0[7];
    lo = lo > 0.f ? lo : 0.f;
    hi = hi > 0.f ? hi : 0.f;
    Ap[id] = f2bf(lo) | (f2bf(hi) << 16);
}

// ---- layer 2: WMMA bf16 GEMM, split-K=2, 2 m-tiles x 2 n-tiles per wave ---
// waves = 8*64*2 = 1024 (128 blocks x 8 waves); 4 independent wmma chains.
// Raw f32 partials (bias/relu folded into layer 3).
__global__ void layer2_wmma_kernel(const unsigned* __restrict__ Ap,
                                   const unsigned* __restrict__ Bp,
                                   float* __restrict__ h2p) {
    int tid  = blockIdx.x * 256 + threadIdx.x;
    int wave = tid >> 5;
    int lane = tid & 31;
    int kh = wave & 1;                // K half: kblk in [kh*64, kh*64+64)
    int ni = (wave >> 1) & 63;        // n-tile pair index
    int mi = wave >> 7;               // m-tile pair index, 0..7
    int m0 = mi * 2, m1 = m0 + 1;
    int n0 = ni * 2, n1 = n0 + 1;

    const unsigned* a0B = Ap + (size_t)(m0 * KBLKS + kh * 64) * 256 + lane * 8;
    const unsigned* a1B = Ap + (size_t)(m1 * KBLKS + kh * 64) * 256 + lane * 8;
    const unsigned* b0B = Bp + (size_t)(n0 * KBLKS + kh * 64) * 256 + lane * 8;
    const unsigned* b1B = Bp + (size_t)(n1 * KBLKS + kh * 64) * 256 + lane * 8;

    v8f acc00 = {}; v8f acc01 = {}; v8f acc10 = {}; v8f acc11 = {};
    for (int kb = 0; kb < 64; ++kb) {
        v16bf a0 = *(const v16bf*)(a0B + kb * 256);
        v16bf a1 = *(const v16bf*)(a1B + kb * 256);
        v16bf b0 = *(const v16bf*)(b0B + kb * 256);
        v16bf b1 = *(const v16bf*)(b1B + kb * 256);
        acc00 = __builtin_amdgcn_wmma_f32_16x16x32_bf16(
                    false, a0, false, b0, (short)0, acc00, false, false);
        acc01 = __builtin_amdgcn_wmma_f32_16x16x32_bf16(
                    false, a0, false, b1, (short)0, acc01, false, false);
        acc10 = __builtin_amdgcn_wmma_f32_16x16x32_bf16(
                    false, a1, false, b0, (short)0, acc10, false, false);
        acc11 = __builtin_amdgcn_wmma_f32_16x16x32_bf16(
                    false, a1, false, b1, (short)0, acc11, false, false);
    }
    // C/D layout scatter: row = m_tile*16 + (lane<16 ? r : 8+r), col = tile*16 + lane%16
    int colin  = lane & 15;
    int rowOff = (lane >> 4) ? 8 : 0;
    int col0 = n0 * 16 + colin, col1 = n1 * 16 + colin;
    float* outp = h2p + (size_t)kh * (B_SZ * H2);
    #pragma unroll
    for (int r = 0; r < 8; ++r) {
        int row0 = m0 * 16 + rowOff + r;
        int row1 = m1 * 16 + rowOff + r;
        outp[(size_t)row0 * H2 + col0] = acc00[r];
        outp[(size_t)row0 * H2 + col1] = acc01[r];
        outp[(size_t)row1 * H2 + col0] = acc10[r];
        outp[(size_t)row1 * H2 + col1] = acc11[r];
    }
}

// ---- layer 3: k = tanh(relu(p0+p1+b2) @ W3^T + b3) + poly(z) --------------
__global__ void layer3_kernel(const float* __restrict__ h2p,
                              const float* __restrict__ b2,
                              const float* __restrict__ W3,
                              const float* __restrict__ b3,
                              const float* __restrict__ z,
                              const float* __restrict__ wp,
                              float* __restrict__ kout) {
    __shared__ float red[256];
    int b = blockIdx.x >> 2;
    int j = blockIdx.x & 3;
    int t = threadIdx.x;
    const float* p0 = h2p + (size_t)b * H2;
    const float* p1 = h2p + (size_t)(B_SZ * H2) + (size_t)b * H2;
    const float* w3 = W3 + (size_t)j * H2;
    float s = 0.f;
    for (int i = t; i < H2; i += 256) {
        float p = p0[i] + p1[i] + b2[i];
        p = p > 0.f ? p : 0.f;
        s += p * w3[i];
    }
    red[t] = s;
    __syncthreads();
    for (int off = 128; off > 0; off >>= 1) {
        if (t < off) red[t] += red[t + off];
        __syncthreads();
    }
    if (t == 0) {
        float val = tanhf(red[0] + b3[j]);
        float A = z[b * 4 + 0], T = z[b * 4 + 1], N = z[b * 4 + 2], C = z[b * 4 + 3];
        float f;
        if      (j == 0) f = wp[0]  + wp[1]  * A + wp[2]  * A * A;
        else if (j == 1) f = wp[3]  + wp[4]  * A + wp[5]  * A * A
                           + wp[6]  * T + wp[7]  * T * T + wp[8]  * A * T;
        else if (j == 2) f = wp[9]  + wp[10] * N + wp[11] * N * N
                           + wp[12] * T + wp[13] * T * T + wp[14] * T * N;
        else             f = wp[15] + wp[16] * C + wp[17] * C * C
                           + wp[18] * N + wp[19] * N * N + wp[20] * N * C;
        kout[b * 4 + j] = val + f;
    }
}

// ---- RK4 stage input: z = y + c*h*k ---------------------------------------
__global__ void stage_update_kernel(const float* __restrict__ y,
                                    const float* __restrict__ k,
                                    const float* __restrict__ s_grid, int t,
                                    float c, float* __restrict__ z) {
    int i = blockIdx.x * 256 + threadIdx.x;
    if (i < B_SZ * D_IN) {
        float h = s_grid[t + 1] - s_grid[t];
        z[i] = y[i] + c * h * k[i];
    }
}

// ---- RK4 combine: y += h/6*(k1+2k2+2k3+k4); z = y; out[t+1] = y -----------
__global__ void final_update_kernel(float* __restrict__ y,
                                    const float* __restrict__ ks,
                                    const float* __restrict__ s_grid, int t,
                                    float* __restrict__ z,
                                    float* __restrict__ out) {
    int i = blockIdx.x * 256 + threadIdx.x;
    if (i < B_SZ * D_IN) {
        float h  = s_grid[t + 1] - s_grid[t];
        float yn = y[i] + (h / 6.f) * (ks[i] + 2.f * ks[1024 + i]
                                     + 2.f * ks[2048 + i] + ks[3072 + i]);
        y[i] = yn; z[i] = yn;
        out[(size_t)(t + 1) * (B_SZ * D_IN) + i] = yn;
    }
}

extern "C" void kernel_launch(void* const* d_in, const int* in_sizes, int n_in,
                              void* d_out, int out_size, void* d_ws, size_t ws_size,
                              hipStream_t stream) {
    const float* s_grid = (const float*)d_in[0];
    const float* y0     = (const float*)d_in[1];
    const float* W1     = (const float*)d_in[2];
    const float* b1     = (const float*)d_in[3];
    const float* W2     = (const float*)d_in[4];
    const float* b2     = (const float*)d_in[5];
    const float* W3     = (const float*)d_in[6];
    const float* b3     = (const float*)d_in[7];
    const float* wpoly  = (const float*)d_in[8];
    float* out = (float*)d_out;

    // workspace layout (bytes)
    char* ws = (char*)d_ws;
    float*    y      = (float*)(ws);                        //   4 KB
    float*    z      = (float*)(ws + 4096);                 //   4 KB
    float*    kst    = (float*)(ws + 8192);                 //  16 KB (4 stages)
    float*    h2p    = (float*)(ws + 32768);                //   4 MB (2 K-partials)
    unsigned* Apack  = (unsigned*)(ws + 32768 + 4194304);   //   2 MB
    unsigned* W2pack = (unsigned*)(ws + 32768 + 6291456);   //  16 MB

    pack_w2_kernel<<<16384, 256, 0, stream>>>(W2, W2pack);
    init_kernel<<<4, 256, 0, stream>>>(y0, y, z, out);

    static const float cstage[3] = {0.5f, 0.5f, 1.0f};
    for (int t = 0; t < NSTEP; ++t) {
        for (int s = 0; s < 4; ++s) {
            layer1_kernel<<<2048, 256, 0, stream>>>(z, W1, b1, Apack);
            layer2_wmma_kernel<<<128, 256, 0, stream>>>(Apack, W2pack, h2p);
            layer3_kernel<<<1024, 256, 0, stream>>>(h2p, b2, W3, b3, z, wpoly,
                                                    kst + s * 1024);
            if (s < 3)
                stage_update_kernel<<<4, 256, 0, stream>>>(y, kst + s * 1024,
                                                           s_grid, t, cstage[s], z);
        }
        final_update_kernel<<<4, 256, 0, stream>>>(y, kst, s_grid, t, z, out);
    }
}
